// DS1_61632780697812
// MI455X (gfx1250) — matrix operand order repaired and checked by reading, not compiled, not run on hardware
//
#include <hip/hip_runtime.h>

#define N_ROWS 16384
#define U_COLS 4096
#define D_DIM  1024
#define KSTEP  64
#define NSTAGE (D_DIM / KSTEP)     // 16
#define LDS_STRIDE 72              // halfs per LDS row (64 payload + 8 pad) = 144B, conflict-free

typedef __attribute__((ext_vector_type(16))) _Float16 v16h;
typedef __attribute__((ext_vector_type(8)))  _Float16 v8h;
typedef __attribute__((ext_vector_type(4)))  _Float16 h4;
typedef __attribute__((ext_vector_type(8)))  float    v8f;
typedef __attribute__((ext_vector_type(4)))  int      v4i;

#define GAS __attribute__((address_space(1)))
#define LAS __attribute__((address_space(3)))

typedef GAS v4i* gv4i_p;   // global int4*
typedef LAS v4i* lv4i_p;   // LDS int4*

#if defined(__has_builtin)
#if __has_builtin(__builtin_amdgcn_global_load_async_to_lds_b128) && \
    __has_builtin(__builtin_amdgcn_s_wait_asynccnt)
#define HAVE_ASYNC_LDS 1
#else
#define HAVE_ASYNC_LDS 0
#endif
#else
#define HAVE_ASYNC_LDS 0
#endif

union Frag16 { v16h v; v8h h[2]; };

// One block per row: convert f32 row -> f16 copy, and emit sum of squares.
__launch_bounds__(256)
__global__ void prep_rows(const float* __restrict__ src,
                          _Float16* __restrict__ dst,
                          float* __restrict__ sq) {
  const int row = blockIdx.x;
  const int t   = threadIdx.x;                       // 256 threads * 4 elems = 1024
  const float4 x = ((const float4*)(src + (size_t)row * D_DIM))[t];
  h4 hv;
  hv.x = (_Float16)x.x; hv.y = (_Float16)x.y;
  hv.z = (_Float16)x.z; hv.w = (_Float16)x.w;
  ((h4*)(dst + (size_t)row * D_DIM))[t] = hv;
  float s = x.x * x.x + x.y * x.y + x.z * x.z + x.w * x.w;
  __shared__ float red[256];
  red[t] = s;
  __syncthreads();
  #pragma unroll
  for (int off = 128; off > 0; off >>= 1) {
    if (t < off) red[t] += red[t + off];
    __syncthreads();
  }
  if (t == 0) sq[row] = red[0];
}

// Copy one 128-row x KSTEP-half tile (global row stride D_DIM) into LDS
// (row stride LDS_STRIDE). 8 waves cooperate: wave wid covers rows wid*16..wid*16+15.
// Each lane moves one 16B chunk; 8 chunks/row -> 4 rows per async instruction.
__device__ __forceinline__ void copy_tile(const _Float16* __restrict__ g0,
                                          _Float16* l0, int wid, int lane) {
  const int chunk = lane & 7;      // which 16B chunk within the row
  const int rsub  = lane >> 3;     // 4 rows per instruction
  #pragma unroll
  for (int t = 0; t < 4; ++t) {
    const int row = wid * 16 + t * 4 + rsub;
    const _Float16* g = g0 + (size_t)row * D_DIM + chunk * 8;
    _Float16* l = l0 + row * LDS_STRIDE + chunk * 8;
#if HAVE_ASYNC_LDS
    __builtin_amdgcn_global_load_async_to_lds_b128(
        (gv4i_p)(uintptr_t)g, (lv4i_p)(uint32_t)(uintptr_t)l,
        /*offset=*/0, /*cpol=*/0);
#else
    *(v8h*)l = *(const v8h*)g;
#endif
  }
}

__device__ __forceinline__ void async_wait_all() {
#if HAVE_ASYNC_LDS
  __builtin_amdgcn_s_wait_asynccnt(0);
#endif
}

// out[n,u] = xsq[n] + wsq[u] - 2 * sum_k Xh[n,k]*Wh[u,k]
// Block tile 128x128, 8 wave32 waves (2x4), each wave 64x32 = 4x2 WMMA tiles.
// K pipelined in stages of 64 through double-buffered LDS via async-to-LDS copies.
__launch_bounds__(256)
__global__ void dist_gemm(const _Float16* __restrict__ Xh,
                          const _Float16* __restrict__ Wh,
                          const float* __restrict__ xsq,
                          const float* __restrict__ wsq,
                          float* __restrict__ out) {
  __shared__ _Float16 lds[2][2][128 * LDS_STRIDE];   // [buf][A/B][...], 72 KB total

  const int lane = threadIdx.x & 31;
  const int wid  = threadIdx.x >> 5;          // 0..7
  const int wm   = wid >> 2;                  // 0..1  -> 64 rows
  const int wn   = wid & 3;                   // 0..3  -> 32 cols
  const int lid  = lane & 15;
  const int hi   = lane >> 4;                 // K-half selector per ISA layout

  const int blockRow = blockIdx.y * 128;
  const int blockCol = blockIdx.x * 128;

  v8f acc[4][2];
  #pragma unroll
  for (int i = 0; i < 4; ++i)
    #pragma unroll
    for (int j = 0; j < 2; ++j) { v8f z = {}; acc[i][j] = z; }

  const _Float16* Ag = Xh + (size_t)blockRow * D_DIM;
  const _Float16* Bg = Wh + (size_t)blockCol * D_DIM;

  // Prologue: stage 0 copies in flight.
  copy_tile(Ag, lds[0][0], wid, lane);
  copy_tile(Bg, lds[0][1], wid, lane);

  for (int s = 0; s < NSTAGE; ++s) {
    const int cur = s & 1;
    async_wait_all();       // our stage-s copies landed in LDS
    __syncthreads();        // everyone's stage-s data ready; everyone done reading buf^1
    if (s + 1 < NSTAGE) {   // overlap next-stage copy with this stage's WMMAs
      copy_tile(Ag + (s + 1) * KSTEP, lds[cur ^ 1][0], wid, lane);
      copy_tile(Bg + (s + 1) * KSTEP, lds[cur ^ 1][1], wid, lane);
    }
    const _Float16* As = lds[cur][0];
    const _Float16* Bs = lds[cur][1];
    #pragma unroll
    for (int kc = 0; kc < 2; ++kc) {          // two 32-wide K chunks per stage
      Frag16 a[4], b[2];
      #pragma unroll
      for (int i = 0; i < 4; ++i) {
        const _Float16* p = As + (wm * 64 + i * 16 + lid) * LDS_STRIDE + kc * 32 + hi * 8;
        a[i].h[0] = *(const v8h*)p;           // K = +hi*8    .. +7
        a[i].h[1] = *(const v8h*)(p + 16);    // K = +16+hi*8 .. +7
      }
      #pragma unroll
      for (int j = 0; j < 2; ++j) {
        const _Float16* p = Bs + (wn * 32 + j * 16 + lid) * LDS_STRIDE + kc * 32 + hi * 8;
        b[j].h[0] = *(const v8h*)p;           // K = +hi*8    .. +7
        b[j].h[1] = *(const v8h*)(p + 16);    // K = +16+hi*8 .. +7
      }
      #pragma unroll
      for (int i = 0; i < 4; ++i)
        #pragma unroll
        for (int j = 0; j < 2; ++j)
          acc[i][j] = __builtin_amdgcn_wmma_f32_16x16x32_f16(
              false, a[i].v, false, b[j].v,
              (short)0, acc[i][j], false, false);
    }
  }

  // Epilogue: C/D layout -> VGPR r, lanes 0-15: M=r, lanes 16-31: M=r+8; N=lid.
  #pragma unroll
  for (int j = 0; j < 2; ++j) {
    const int col = blockCol + wn * 32 + j * 16 + lid;
    const float wq = wsq[col];
    #pragma unroll
    for (int i = 0; i < 4; ++i) {
      const int r0 = blockRow + wm * 64 + i * 16 + hi * 8;
      float* o = out + (size_t)r0 * U_COLS + col;
      #pragma unroll
      for (int r = 0; r < 8; ++r)
        o[(size_t)r * U_COLS] = xsq[r0 + r] + wq - 2.0f * acc[i][j][r];
    }
  }
}

extern "C" void kernel_launch(void* const* d_in, const int* in_sizes, int n_in,
                              void* d_out, int out_size, void* d_ws, size_t ws_size,
                              hipStream_t stream) {
  const float* x = (const float*)d_in[0];   // [N, D] f32
  const float* w = (const float*)d_in[1];   // [U, D] f32
  float* out = (float*)d_out;               // [N, U] f32

  char* ws = (char*)d_ws;
  _Float16* Xh = (_Float16*)ws;                                  // 32 MB
  _Float16* Wh = (_Float16*)(ws + (size_t)N_ROWS * D_DIM * 2);   //  8 MB
  float* xsq = (float*)(ws + (size_t)N_ROWS * D_DIM * 2
                            + (size_t)U_COLS * D_DIM * 2);       // 64 KB
  float* wsq = xsq + N_ROWS;                                     // 16 KB

  prep_rows<<<N_ROWS, 256, 0, stream>>>(x, Xh, xsq);
  prep_rows<<<U_COLS, 256, 0, stream>>>(w, Wh, wsq);
  dist_gemm<<<dim3(U_COLS / 128, N_ROWS / 128), 256, 0, stream>>>(Xh, Wh, xsq, wsq, out);
}
